// DIENCollaborative_77034533421638
// MI455X (gfx1250) — matrix operand order repaired and checked by reading, not compiled, not run on hardware
//
#include <hip/hip_runtime.h>
#include <hip/hip_bf16.h>
#include <stdint.h>

// ---------------------------------------------------------------------------
// DIEN fused kernel for MI455X (gfx1250, wave32, WMMA bf16 16x16x32).
// One wave owns a 16-row batch tile and runs GRU + attention + AUGRU in
// lockstep over T=200 steps with all weights resident in LDS as bf16.
// item_emb (25.6MB) is L2-resident; the seq_emb gather is software-pipelined
// one step ahead so global-load latency overlaps the 96 WMMAs per step.
// target_proj, attention weights and biases are register-resident, so the
// only LDS traffic inside the scan is weight B-fragments + the h bounce.
// ---------------------------------------------------------------------------

#define B_TOTAL 4096
#define T_SEQ   200
#define DD      64
#define G3      192      // 3*D gate rows
#define WSTR    72       // padded LDS row stride (bf16 elems) to spread banks
#define WAVES   2        // waves per block (share the LDS weight copy)
#define BM      16       // batch rows per wave (WMMA M)
#define NTHREADS (WAVES * 32)

typedef __bf16 bf16_t;
typedef __attribute__((ext_vector_type(16))) __bf16 bf16x16;
typedef __attribute__((ext_vector_type(8)))  float  f32x8;
typedef __attribute__((ext_vector_type(4)))  unsigned int u32x4;
typedef struct { u32x4 a; u32x4 b; } u32x4x2;

static __device__ __forceinline__ f32x8 wmma_bf16(bf16x16 a, bf16x16 b, f32x8 c) {
  // D = A(16x32 bf16) * B(32x16 bf16) + C(16x16 f32)
  return __builtin_amdgcn_wmma_f32_16x16x32_bf16(false, a, false, b, (short)0, c,
                                                 false, false);
}

static __device__ __forceinline__ f32x8 bias8(float v) {
  f32x8 c;
#pragma unroll
  for (int i = 0; i < 8; ++i) c[i] = v;
  return c;
}

static __device__ __forceinline__ float sigmoidf1(float x) {
  return 1.0f / (1.0f + __expf(-x));
}
static __device__ __forceinline__ float tanhf1(float x) {
  float e = __expf(2.0f * x);
  return 1.0f - 2.0f / (e + 1.0f);   // safe at +/-inf
}

// Load 16 bf16 from two 16-byte-aligned LDS runs into a v16bf fragment.
static __device__ __forceinline__ bf16x16 load2x16(const bf16_t* p0, const bf16_t* p1) {
  u32x4x2 t;
  t.a = *(const u32x4*)p0;
  t.b = *(const u32x4*)p1;
  return __builtin_bit_cast(bf16x16, t);
}

// B fragment (32x16 bf16) from LDS weight matrix stored [N][K] row-major
// (row stride WSTR).  B[k][n] = W[nbase+n][kbase+k].  Assumed lane layout:
// lane holds column n = lane&15; lanes 0-15 carry K=kbase+{0..7,16..23},
// lanes 16-31 carry K=kbase+{8..15,24..31} (matches the A fragment).
static __device__ __forceinline__ bf16x16 load_b_frag(const bf16_t* W, int nbase,
                                                      int kbase, int lane) {
  const bf16_t* p = W + (nbase + (lane & 15)) * WSTR + kbase + ((lane >> 4) << 3);
  return load2x16(p, p + 16);
}

// A fragment (16x32 bf16) from an LDS tile [16][WSTR].  ISA layout:
// lanes 0-15: M=lane, K = ks*32 + {0..7, 16..23};
// lanes16-31: M=lane-16, K = ks*32 + {8..15, 24..31}.
static __device__ __forceinline__ bf16x16 load_a_frag_lds(const bf16_t* H, int ks,
                                                          int lane) {
  const bf16_t* row = H + (lane & 15) * WSTR;
  int h8 = (lane >> 4) << 3;
  return load2x16(row + ks * 32 + h8, row + ks * 32 + 16 + h8);
}

// Raw per-lane slice of one global f32 row (the 32 elems this lane feeds into
// its two A fragments).  Kept in registers for software pipelining.
struct RawRow { float4 q[8]; };

static __device__ __forceinline__ RawRow load_raw_row(const float* row, int lane) {
  const int h8 = (lane >> 4) << 3;
  RawRow r;
#pragma unroll
  for (int ks = 0; ks < 2; ++ks) {
    const int k0 = ks * 32 + h8;
    r.q[ks * 4 + 0] = *(const float4*)(row + k0);
    r.q[ks * 4 + 1] = *(const float4*)(row + k0 + 4);
    r.q[ks * 4 + 2] = *(const float4*)(row + k0 + 16);
    r.q[ks * 4 + 3] = *(const float4*)(row + k0 + 20);
  }
  return r;
}

static __device__ __forceinline__ bf16x16 pack_frag(const RawRow& rr, int ks) {
  bf16x16 v;
#pragma unroll
  for (int j = 0; j < 4; ++j) {
    const float4 f = rr.q[ks * 4 + j];
    v[j * 4 + 0] = (bf16_t)f.x;
    v[j * 4 + 1] = (bf16_t)f.y;
    v[j * 4 + 2] = (bf16_t)f.z;
    v[j * 4 + 3] = (bf16_t)f.w;
  }
  return v;
}

// Sum across the 16 lanes of a half-wave (xor masks 1,2,4,8 keep bit4).
static __device__ __forceinline__ float reduce16(float v) {
  v += __shfl_xor(v, 1, 32);
  v += __shfl_xor(v, 2, 32);
  v += __shfl_xor(v, 4, 32);
  v += __shfl_xor(v, 8, 32);
  return v;
}

// Dynamic LDS footprint (bytes): 4 weight matrices + 2 h tiles per wave.
#define SHMEM_BYTES \
  ((size_t)4 * G3 * WSTR * 2 + (size_t)2 * WAVES * BM * WSTR * 2)

__global__ __launch_bounds__(NTHREADS) void dien_fused_kernel(
    const int* __restrict__ u_idx, const int* __restrict__ i_idx,
    const int* __restrict__ seq, const float* __restrict__ item_emb,
    const float* __restrict__ gru_w_ih, const float* __restrict__ gru_b_ih,
    const float* __restrict__ gru_w_hh, const float* __restrict__ gru_b_hh,
    const float* __restrict__ attn_w, const float* __restrict__ attn_b,
    const float* __restrict__ wr_w, const float* __restrict__ wr_b,
    const float* __restrict__ wz_w, const float* __restrict__ wz_b,
    const float* __restrict__ wh_w, const float* __restrict__ wh_b,
    const float* __restrict__ aux_w, const float* __restrict__ aux_b,
    float* __restrict__ out) {
  extern __shared__ char smem[];
  bf16_t* WIH = (bf16_t*)smem;               // [G3][WSTR]  gru_w_ih (bf16)
  bf16_t* WHH = WIH + G3 * WSTR;             // [G3][WSTR]  gru_w_hh
  bf16_t* WXA = WHH + G3 * WSTR;             // [G3][WSTR]  wr_x|wz_x|wh_x
  bf16_t* WHA = WXA + G3 * WSTR;             // [G3][WSTR]  wr_h|wz_h|wh_h
  bf16_t* HG  = WHA + G3 * WSTR;             // [WAVES][BM][WSTR] GRU h
  bf16_t* HAT = HG + WAVES * BM * WSTR;      // [WAVES][BM][WSTR] AUGRU h

  const int tid  = threadIdx.x;
  const int lane = tid & 31;
  const int wave = tid >> 5;
  const int n    = lane & 15;   // local column within a 16-wide N slab
  const int half = lane >> 4;   // row-group selector for C/D fragments
  const int m0   = (blockIdx.x * WAVES + wave) * BM;

  // ---- Phase 0: stage attn_w (bf16) into the WIH region ----
  for (int i = tid; i < DD * DD; i += NTHREADS)
    WIH[(i / DD) * WSTR + (i % DD)] = (bf16_t)attn_w[i];
  __syncthreads();

  bf16_t* hg = HG  + wave * BM * WSTR;
  bf16_t* ha = HAT + wave * BM * WSTR;

  // ---- target_proj = item_emb[i_idx] @ attn_w^T + attn_b ----
  // Produced directly in C-fragment layout; lives in registers for the scan.
  f32x8 tpv[4];
  {
    const float* trow = item_emb + (size_t)i_idx[m0 + n] * DD;
    RawRow traw = load_raw_row(trow, lane);
    bf16x16 at0 = pack_frag(traw, 0);
    bf16x16 at1 = pack_frag(traw, 1);
#pragma unroll
    for (int ns = 0; ns < 4; ++ns) {
      f32x8 c = bias8(attn_b[ns * 16 + n]);
      c = wmma_bf16(at0, load_b_frag(WIH, ns * 16, 0, lane), c);
      c = wmma_bf16(at1, load_b_frag(WIH, ns * 16, 32, lane), c);
      tpv[ns] = c;
    }
  }
  __syncthreads();

  // ---- Load the four 192x64 weight matrices as bf16 with padded rows ----
  for (int i = tid; i < G3 * DD; i += NTHREADS) {
    int g = i / DD, d = i % DD;
    WIH[g * WSTR + d] = (bf16_t)gru_w_ih[i];
    WHH[g * WSTR + d] = (bf16_t)gru_w_hh[i];
  }
  for (int i = tid; i < DD * DD; i += NTHREADS) {
    int r = i / DD, d = i % DD;
    WXA[(0 * DD + r) * WSTR + d] = (bf16_t)wr_w[r * 2 * DD + d];
    WXA[(1 * DD + r) * WSTR + d] = (bf16_t)wz_w[r * 2 * DD + d];
    WXA[(2 * DD + r) * WSTR + d] = (bf16_t)wh_w[r * 2 * DD + d];
    WHA[(0 * DD + r) * WSTR + d] = (bf16_t)wr_w[r * 2 * DD + DD + d];
    WHA[(1 * DD + r) * WSTR + d] = (bf16_t)wz_w[r * 2 * DD + DD + d];
    WHA[(2 * DD + r) * WSTR + d] = (bf16_t)wh_w[r * 2 * DD + DD + d];
  }
  for (int i = lane; i < BM * WSTR; i += 32) {
    hg[i] = (bf16_t)0.0f;
    ha[i] = (bf16_t)0.0f;
  }

  // ---- Loop-invariant biases -> registers (read once from global/L2) ----
  float bgr[4], bgz[4], bxn[4], bhn[4], bar_[4], baz_[4], bah_[4];
#pragma unroll
  for (int ns = 0; ns < 4; ++ns) {
    const int nc = ns * 16 + n;
    bgr[ns]  = gru_b_ih[nc] + gru_b_hh[nc];
    bgz[ns]  = gru_b_ih[DD + nc] + gru_b_hh[DD + nc];
    bxn[ns]  = gru_b_ih[2 * DD + nc];
    bhn[ns]  = gru_b_hh[2 * DD + nc];
    bar_[ns] = wr_b[nc];
    baz_[ns] = wz_b[nc];
    bah_[ns] = wh_b[nc];
  }
  __syncthreads();

  // ---- Software-pipelined scan: gather for t+1 overlaps compute for t ----
  const int seq_base = (m0 + n) * T_SEQ;
  RawRow cur = load_raw_row(item_emb + (size_t)seq[seq_base] * DD, lane);

  for (int t = 0; t < T_SEQ; ++t) {
    bf16x16 ax0 = pack_frag(cur, 0);
    bf16x16 ax1 = pack_frag(cur, 1);

    // Issue next step's gather now; it completes during the 96 WMMAs below.
    const int tn = (t + 1 < T_SEQ) ? t + 1 : t;
    RawRow nxt = load_raw_row(item_emb + (size_t)seq[seq_base + tn] * DD, lane);

    bf16x16 ah0 = load_a_frag_lds(hg, 0, lane);
    bf16x16 ah1 = load_a_frag_lds(hg, 1, lane);

    float pr[8];
#pragma unroll
    for (int r = 0; r < 8; ++r) pr[r] = 0.0f;

    // ================= GRU step =================
#pragma unroll
    for (int ns = 0; ns < 4; ++ns) {
      const int nc = ns * 16 + n;
      f32x8 cr = bias8(bgr[ns]);
      cr = wmma_bf16(ax0, load_b_frag(WIH, ns * 16, 0, lane), cr);
      cr = wmma_bf16(ax1, load_b_frag(WIH, ns * 16, 32, lane), cr);
      cr = wmma_bf16(ah0, load_b_frag(WHH, ns * 16, 0, lane), cr);
      cr = wmma_bf16(ah1, load_b_frag(WHH, ns * 16, 32, lane), cr);
      f32x8 cz = bias8(bgz[ns]);
      cz = wmma_bf16(ax0, load_b_frag(WIH, DD + ns * 16, 0, lane), cz);
      cz = wmma_bf16(ax1, load_b_frag(WIH, DD + ns * 16, 32, lane), cz);
      cz = wmma_bf16(ah0, load_b_frag(WHH, DD + ns * 16, 0, lane), cz);
      cz = wmma_bf16(ah1, load_b_frag(WHH, DD + ns * 16, 32, lane), cz);
      f32x8 cx = bias8(bxn[ns]);  // xn (x and h parts kept separate)
      cx = wmma_bf16(ax0, load_b_frag(WIH, 2 * DD + ns * 16, 0, lane), cx);
      cx = wmma_bf16(ax1, load_b_frag(WIH, 2 * DD + ns * 16, 32, lane), cx);
      f32x8 ch = bias8(bhn[ns]);  // hn
      ch = wmma_bf16(ah0, load_b_frag(WHH, 2 * DD + ns * 16, 0, lane), ch);
      ch = wmma_bf16(ah1, load_b_frag(WHH, 2 * DD + ns * 16, 32, lane), ch);
#pragma unroll
      for (int r = 0; r < 8; ++r) {
        const int Mr = r + 8 * half;
        const float rg   = sigmoidf1(cr[r]);
        const float zg   = sigmoidf1(cz[r]);
        const float ng   = tanhf1(cx[r] + rg * ch[r]);
        const float hold = (float)hg[Mr * WSTR + nc];
        const float hnew = (1.0f - zg) * ng + zg * hold;
        hg[Mr * WSTR + nc] = (bf16_t)hnew;
        pr[r] += hnew * tpv[ns][r];  // attention dot partial (registers)
      }
    }

    // a_t = sigmoid(gru_out_t . target_proj) — reduce16 leaves the value in
    // every lane of the half that owns rows r+8*half, so no LDS round-trip.
    float av[8];
#pragma unroll
    for (int r = 0; r < 8; ++r) av[r] = sigmoidf1(reduce16(pr[r]));

    // ================= AUGRU step (uses a_t immediately) =================
    bf16x16 aa0 = load_a_frag_lds(ha, 0, lane);
    bf16x16 aa1 = load_a_frag_lds(ha, 1, lane);

#pragma unroll
    for (int ns = 0; ns < 4; ++ns) {
      const int nc = ns * 16 + n;
      f32x8 cr = bias8(bar_[ns]);
      cr = wmma_bf16(ax0, load_b_frag(WXA, ns * 16, 0, lane), cr);
      cr = wmma_bf16(ax1, load_b_frag(WXA, ns * 16, 32, lane), cr);
      cr = wmma_bf16(aa0, load_b_frag(WHA, ns * 16, 0, lane), cr);
      cr = wmma_bf16(aa1, load_b_frag(WHA, ns * 16, 32, lane), cr);
      f32x8 cz = bias8(baz_[ns]);
      cz = wmma_bf16(ax0, load_b_frag(WXA, DD + ns * 16, 0, lane), cz);
      cz = wmma_bf16(ax1, load_b_frag(WXA, DD + ns * 16, 32, lane), cz);
      cz = wmma_bf16(aa0, load_b_frag(WHA, DD + ns * 16, 0, lane), cz);
      cz = wmma_bf16(aa1, load_b_frag(WHA, DD + ns * 16, 32, lane), cz);
      f32x8 cx = bias8(bah_[ns]);  // xh (includes wh_b)
      cx = wmma_bf16(ax0, load_b_frag(WXA, 2 * DD + ns * 16, 0, lane), cx);
      cx = wmma_bf16(ax1, load_b_frag(WXA, 2 * DD + ns * 16, 32, lane), cx);
      f32x8 ch = bias8(0.0f);      // h @ wh_h^T (no bias)
      ch = wmma_bf16(aa0, load_b_frag(WHA, 2 * DD + ns * 16, 0, lane), ch);
      ch = wmma_bf16(aa1, load_b_frag(WHA, 2 * DD + ns * 16, 32, lane), ch);
#pragma unroll
      for (int r = 0; r < 8; ++r) {
        const int Mr = r + 8 * half;
        const float rg   = sigmoidf1(cr[r]);
        const float zg   = av[r] * sigmoidf1(cz[r]);
        const float hh   = tanhf1(cx[r] + rg * ch[r]);
        const float hold = (float)ha[Mr * WSTR + nc];
        const float hnew = (1.0f - zg) * hold + zg * hh;
        ha[Mr * WSTR + nc] = (bf16_t)hnew;
      }
    }

    cur = nxt;
  }

  // ---- attn_vec output (coalesced 16-lane rows) ----
#pragma unroll
  for (int ns = 0; ns < 4; ++ns)
#pragma unroll
    for (int r = 0; r < 8; ++r)
      out[B_TOTAL + (m0 + r + 8 * half) * DD + ns * 16 + n] =
          (float)ha[(r + 8 * half) * WSTR + ns * 16 + n];

  // ---- aux_logits from the final GRU hidden state ----
  float pa[8];
#pragma unroll
  for (int r = 0; r < 8; ++r) pa[r] = 0.0f;
#pragma unroll
  for (int ns = 0; ns < 4; ++ns) {
    const float aw = aux_w[ns * 16 + n];
#pragma unroll
    for (int r = 0; r < 8; ++r)
      pa[r] += (float)hg[(r + 8 * half) * WSTR + ns * 16 + n] * aw;
  }
#pragma unroll
  for (int r = 0; r < 8; ++r) {
    float s = reduce16(pa[r]) + aux_b[0];
    if (n == r) out[B_TOTAL + B_TOTAL * DD + m0 + r + 8 * half] = s;
  }
}

__global__ void dien_fm1_kernel(const int* __restrict__ u_idx,
                                const int* __restrict__ i_idx,
                                const float* __restrict__ user_bias,
                                const float* __restrict__ item_bias,
                                float* __restrict__ out) {
  int m = blockIdx.x * blockDim.x + threadIdx.x;
  if (m < B_TOTAL) out[m] = user_bias[u_idx[m]] + item_bias[i_idx[m]];
}

extern "C" void kernel_launch(void* const* d_in, const int* in_sizes, int n_in,
                              void* d_out, int out_size, void* d_ws, size_t ws_size,
                              hipStream_t stream) {
  (void)in_sizes; (void)n_in; (void)out_size; (void)d_ws; (void)ws_size;
  const int*   u_idx     = (const int*)d_in[0];
  const int*   i_idx     = (const int*)d_in[1];
  const int*   seq       = (const int*)d_in[2];
  const float* item_emb  = (const float*)d_in[3];
  const float* user_bias = (const float*)d_in[4];
  const float* item_bias = (const float*)d_in[5];
  const float* gru_w_ih  = (const float*)d_in[6];
  const float* gru_b_ih  = (const float*)d_in[7];
  const float* gru_w_hh  = (const float*)d_in[8];
  const float* gru_b_hh  = (const float*)d_in[9];
  const float* attn_w    = (const float*)d_in[10];
  const float* attn_b    = (const float*)d_in[11];
  const float* wr_w      = (const float*)d_in[12];
  const float* wr_b      = (const float*)d_in[13];
  const float* wz_w      = (const float*)d_in[14];
  const float* wz_b      = (const float*)d_in[15];
  const float* wh_w      = (const float*)d_in[16];
  const float* wh_b      = (const float*)d_in[17];
  const float* aux_w     = (const float*)d_in[18];
  const float* aux_b     = (const float*)d_in[19];
  float* out = (float*)d_out;

  // ~117 KB dynamic LDS (gfx1250 WGP has 320 KB) — opt in explicitly.
  (void)hipFuncSetAttribute(reinterpret_cast<const void*>(dien_fused_kernel),
                            hipFuncAttributeMaxDynamicSharedMemorySize,
                            (int)SHMEM_BYTES);

  dim3 grid(B_TOTAL / (BM * WAVES));  // 128 blocks x 64 threads (2 waves)
  dien_fused_kernel<<<grid, NTHREADS, SHMEM_BYTES, stream>>>(
      u_idx, i_idx, seq, item_emb, gru_w_ih, gru_b_ih, gru_w_hh, gru_b_hh,
      attn_w, attn_b, wr_w, wr_b, wz_w, wz_b, wh_w, wh_b, aux_w, aux_b, out);

  dien_fm1_kernel<<<dim3((B_TOTAL + 255) / 256), 256, 0, stream>>>(
      u_idx, i_idx, user_bias, item_bias, out);
}
